// SoftToHardEncoder_14267881357912
// MI455X (gfx1250) — matrix operand order repaired and checked by reading, not compiled, not run on hardware
//
#include <hip/hip_runtime.h>

typedef __attribute__((ext_vector_type(16))) _Float16 v16h;
typedef __attribute__((ext_vector_type(8)))  float    v8f;

#define B_   16
#define C_   32
#define H_   64
#define W_   64
#define K_   64
#define NTOT (B_*C_*H_*W_)   // 2097152 elements per output tensor

// One block per (b,h) row: stages z[b,:,h,:] (32x64 f32) and the codebook
// (32x64 f32) in LDS, computes soft/hard/idx for all 2048 (w,c) elements,
// stages results in LDS, writes back coalesced.
// Each wave processes 16-element groups (one channel c, 16 consecutive w):
//   lane = (M = element 0..15, half = K-half); lane computes the 32
//   |x-w_k| distances + exp2 probs that fall into its WMMA A-fragment slots,
//   then V_WMMA_F32_16X16X32_F16 x2 reduces num = sum(p*w) (B col 0) and
//   den = sum(p) (B col 1) with f32 accumulation.
__global__ __launch_bounds__(256) void s2h_kernel(const float* __restrict__ z,
                                                  const float* __restrict__ codes,
                                                  float* __restrict__ out)
{
    __shared__ float lds_codes[C_ * K_];   // 8 KB, f32 codebook
    __shared__ float lds_z[C_ * W_];       // 8 KB, one (b,h) row, all channels
    __shared__ float lds_xfer[8 * 32];     // per-wave num/den exchange
    __shared__ float lds_soft[C_ * W_];    // staged outputs (w-major, c-minor)
    __shared__ float lds_hard[C_ * W_];
    __shared__ float lds_idx [C_ * W_];

    const int t  = threadIdx.x;
    const int bh = blockIdx.x;          // 0..1023
    const int b  = bh >> 6;
    const int h  = bh & 63;

    // Coalesced staging: codes (2048 f32) and z row tile (2048 f32).
#pragma unroll
    for (int j = 0; j < 8; ++j) {
        const int i = t + j * 256;      // i = c*64 + w
        lds_codes[i] = codes[i];
        const int c = i >> 6, w = i & 63;
        lds_z[i] = z[b * (C_*H_*W_) + c * (H_*W_) + h * W_ + w];
    }
    __syncthreads();

    const int wave = t >> 5;
    const int lane = t & 31;
    const int M    = lane & 15;   // element row (and B/D column id)
    const int half = lane >> 4;   // K-half selector

    for (int cc = 0; cc < 4; ++cc) {
        const int c = (wave << 2) | cc;
        const float* cw = lds_codes + c * K_;

        // Per-lane code values at this lane's A-fragment K slots,
        // and the B fragments (col0 = w_c, col1 = ones, rest 0).
        float cA0[16], cA1[16];
        v16h bf0, bf1;
#pragma unroll
        for (int j = 0; j < 16; ++j) {
            const int kA = j + 8 * ((j >> 3) + half);   // A: 16x32 f16 layout
            cA0[j] = cw[kA];
            cA1[j] = cw[kA + 32];
            const int kB = j + 8 * ((j >> 3) + half);   // B: 32x16 f16 layout
            _Float16 v0 = (_Float16)0.f, v1 = (_Float16)0.f;
            if (M == 0)      { v0 = (_Float16)cw[kB]; v1 = (_Float16)cw[kB + 32]; }
            else if (M == 1) { v0 = (_Float16)1.f;    v1 = (_Float16)1.f; }
            bf0[j] = v0;
            bf1[j] = v1;
        }

#pragma unroll
        for (int wt = 0; wt < 4; ++wt) {
            const float x = lds_z[c * W_ + wt * 16 + M];

            // Distances + lexicographic (d, idx) argmin over this lane's 32 codes.
            float d0[16], d1[16];
            float dmin = 3.0e38f;
            int   imin = 0;
#pragma unroll
            for (int j = 0; j < 16; ++j) {
                const int kA = j + 8 * ((j >> 3) + half);
                d0[j] = __builtin_fabsf(x - cA0[j]);
                d1[j] = __builtin_fabsf(x - cA1[j]);
                if (d0[j] < dmin || (d0[j] == dmin && kA        < imin)) { dmin = d0[j]; imin = kA; }
                if (d1[j] < dmin || (d1[j] == dmin && (kA + 32) < imin)) { dmin = d1[j]; imin = kA + 32; }
            }
            // Combine across the two K-halves (wave32 shuffle).
            const float dmin2 = __shfl_xor(dmin, 16, 32);
            const int   imin2 = __shfl_xor(imin, 16, 32);
            if (dmin2 < dmin || (dmin2 == dmin && imin2 < imin)) { dmin = dmin2; imin = imin2; }

            // Shifted softmax numerators: p = exp(dmin - d) in f16 (<= 1.0).
            v16h a0, a1;
#pragma unroll
            for (int j = 0; j < 16; ++j) {
                a0[j] = (_Float16)__builtin_amdgcn_exp2f((dmin - d0[j]) * 1.44269504088896340736f);
                a1[j] = (_Float16)__builtin_amdgcn_exp2f((dmin - d1[j]) * 1.44269504088896340736f);
            }

            // D = A(16x64 probs) x B(64x2: [w_c, 1]) with f32 accumulate.
            v8f acc = {};
            acc = __builtin_amdgcn_wmma_f32_16x16x32_f16(false, a0, false, bf0,
                                                         (short)0, acc, false, false);
            acc = __builtin_amdgcn_wmma_f32_16x16x32_f16(false, a1, false, bf1,
                                                         (short)0, acc, false, false);

            // Lanes holding D columns 0 (num) / 1 (den) spill their 8 rows;
            // wave-local LDS ops are in-order, fence the compiler + DScnt.
            if (M < 2) {
                float* xs = lds_xfer + wave * 32 + M * 16 + half * 8;
#pragma unroll
                for (int r = 0; r < 8; ++r) xs[r] = acc[r];
            }
            __asm__ volatile("s_wait_dscnt 0" ::: "memory");
            if (half == 0) {
                const float num = lds_xfer[wave * 32 + M];
                const float den = lds_xfer[wave * 32 + 16 + M];
                const int o = (wt * 16 + M) * C_ + c;   // (w, c) flat, NHWC order
                lds_soft[o] = num / den;
                lds_hard[o] = cw[imin];
                lds_idx [o] = (float)imin;              // argmin as f32 (tuple promotion)
            }
            __asm__ volatile("s_wait_dscnt 0" ::: "memory");
        }
    }

    __syncthreads();

    // Coalesced write-back of the three outputs, concatenated in return order.
    const int base = bh * (C_ * W_);
#pragma unroll
    for (int j = 0; j < 8; ++j) {
        const int i = t + j * 256;
        out[             base + i] = lds_soft[i];
        out[    NTOT +   base + i] = lds_hard[i];
        out[2 * NTOT +   base + i] = lds_idx[i];
    }
}

extern "C" void kernel_launch(void* const* d_in, const int* in_sizes, int n_in,
                              void* d_out, int out_size, void* d_ws, size_t ws_size,
                              hipStream_t stream)
{
    const float* z     = (const float*)d_in[0];   // (16,32,64,64) f32
    const float* codes = (const float*)d_in[1];   // (32,64) f32
    float* out = (float*)d_out;                   // soft | hard | idx, 3*2097152 f32
    (void)in_sizes; (void)n_in; (void)out_size; (void)d_ws; (void)ws_size;

    dim3 grid(B_ * H_);   // 1024 blocks, one per (b,h) row
    dim3 block(256);      // 8 wave32 waves
    s2h_kernel<<<grid, block, 0, stream>>>(z, codes, out);
}
